// MultiScaleMaskedTransformerDecoder_47459388621436
// MI455X (gfx1250) — matrix operand order repaired and checked by reading, not compiled, not run on hardware
//
#include <hip/hip_runtime.h>
#include <hip/hip_bf16.h>

// ---------------------------------------------------------------------------
// MI455X (gfx1250) implementation: all GEMMs via v_wmma_f32_16x16x32_f16
// (f16 inputs, f32 accumulate), attention fully fused (flash-style) so the
// [B,H,Q,S] score tensor (52MB/layer) never touches memory.
// ---------------------------------------------------------------------------

typedef _Float16 v16h __attribute__((ext_vector_type(16)));
typedef float    v8f  __attribute__((ext_vector_type(8)));
typedef _Float16 h2   __attribute__((ext_vector_type(2)));

union Frag { v16h v; h2 p[8]; };

static __device__ __forceinline__ v8f v8f_zero() {
  v8f z = {0.f, 0.f, 0.f, 0.f, 0.f, 0.f, 0.f, 0.f};
  return z;
}
static __device__ __forceinline__ int imin(int a, int b) { return a < b ? a : b; }

// ---------------------------------------------------------------------------
// Generic batched GEMM: O[b] = act( A[b] (MxK) @ W[b] (KxN) + bias )
// Block tile 128(M) x 128(N), BK=32.  8 waves: wave = (mt in 0..3, ng in 0..1);
// each wave owns a 32x64 strip -> 2 A-frags x 4 B-frags -> 8 WMMAs / K-step
// (3 ds_load_2addr per WMMA).  flags bit0 = relu.
// ---------------------------------------------------------------------------
__global__ __launch_bounds__(256) void gemm_kernel(
    const float* __restrict__ A, int lda, long long bsA,
    const float* __restrict__ W, long long bsW,
    const float* __restrict__ bias,
    float* __restrict__ O, int ldo, long long bsO,
    int M, int N, int K, int flags)
{
  A += (size_t)blockIdx.z * bsA;
  W += (size_t)blockIdx.z * bsW;
  O += (size_t)blockIdx.z * bsO;

  const int tid = threadIdx.x, wid = tid >> 5, lane = tid & 31;
  const int lhalf = lane >> 4, lmod = lane & 15;
  const int mt = wid & 3, ng = wid >> 2;

  __shared__ _Float16 sA[128][34];   // [m][k], padded vs bank conflicts
  __shared__ _Float16 sBT[128][34];  // [n][k] (transposed) -> h2 pairs along K

  v8f acc[2][4];
#pragma unroll
  for (int mi = 0; mi < 2; ++mi)
#pragma unroll
    for (int nt = 0; nt < 4; ++nt) acc[mi][nt] = v8f_zero();

  // staging assignments (fixed per thread)
  const int rA = tid >> 1, cA = (tid & 1) * 16;              // A: 128 rows x 32 cols
  const int gmA = imin((int)blockIdx.y * 128 + rA, M - 1);   // clamp (M tail tiles)
  const float* aptr = A + (size_t)gmA * lda + cA;
  const int rB = tid >> 3, cB = (tid & 7) * 16;              // W: 32 rows x 128 cols
  const float* wptr = W + (size_t)rB * N + (size_t)blockIdx.x * 128 + cB;

  for (int kb = 0; kb < K; kb += 32) {
    __syncthreads();
#pragma unroll
    for (int u = 0; u < 16; ++u) sA[rA][cA + u] = (_Float16)aptr[kb + u];
#pragma unroll
    for (int u = 0; u < 16; ++u) sBT[cB + u][rB] = (_Float16)wptr[(size_t)kb * N + u];
    if (kb + 32 < K) {  // gfx1250 global_prefetch of next K tiles
      __builtin_prefetch(aptr + kb + 32, 0, 1);
      __builtin_prefetch(wptr + (size_t)(kb + 32) * N, 0, 1);
    }
    __syncthreads();

    // A fragments (16x32 f16): lane m = lmod; k0 = 16*(j>>2) + 8*lhalf + 2*(j&3)
    Frag a[2];
#pragma unroll
    for (int mi = 0; mi < 2; ++mi)
#pragma unroll
      for (int j = 0; j < 8; ++j) {
        const int k0 = 16 * (j >> 2) + 8 * lhalf + 2 * (j & 3);
        a[mi].p[j] = *(const h2*)&sA[mt * 32 + mi * 16 + lmod][k0];
      }
#pragma unroll
    for (int nt = 0; nt < 4; ++nt) {
      // B fragment (32x16 f16): lane n = lmod; k pair = 16*(j>>2) + 8*lhalf + 2*(j&3)
      Frag bf;
      const int ncol = ng * 64 + nt * 16 + lmod;
#pragma unroll
      for (int j = 0; j < 8; ++j)
        bf.p[j] = *(const h2*)&sBT[ncol][16 * (j >> 2) + 8 * lhalf + 2 * (j & 3)];
#pragma unroll
      for (int mi = 0; mi < 2; ++mi)
        acc[mi][nt] = __builtin_amdgcn_wmma_f32_16x16x32_f16(
            false, a[mi].v, false, bf.v, (short)0, acc[mi][nt], false, false);
    }
  }

  const bool relu = (flags & 1) != 0;
#pragma unroll
  for (int mi = 0; mi < 2; ++mi)
#pragma unroll
    for (int nt = 0; nt < 4; ++nt) {
#pragma unroll
      for (int r = 0; r < 8; ++r) {
        const int gm = (int)blockIdx.y * 128 + mt * 32 + mi * 16 + r + 8 * lhalf;
        const int gn = (int)blockIdx.x * 128 + ng * 64 + nt * 16 + lmod;
        if (gm < M) {
          float v = acc[mi][nt][r];
          if (bias) v += bias[gn];
          if (relu) v = fmaxf(v, 0.f);
          O[(size_t)gm * ldo + gn] = v;
        }
      }
    }
}

// ---------------------------------------------------------------------------
// Fused multi-head attention with rel-position bias + optional mask bias.
// grid = (H, B); block = 256 (8 waves).  Wave w owns query rows [16w,16w+16).
// Per 64-key chunk: 4 score WMMAs -> online softmax (shfl_xor reductions
// within 16-lane halves) -> P staged to per-wave LDS (C-layout -> A-layout
// conversion) -> 4 P*V WMMAs.  Output written per head into [B,Qn,256].
// ---------------------------------------------------------------------------
__global__ __launch_bounds__(256) void attn_kernel(
    const float* __restrict__ qh, const float* __restrict__ kh,
    const float* __restrict__ vh, const float* __restrict__ rel,
    const float* __restrict__ mbias, float* __restrict__ outp,
    int Qn, int Skv, float scale)
{
  const int h = blockIdx.x, b = blockIdx.y;
  const int tid = threadIdx.x, wid = tid >> 5, lane = tid & 31;
  const int lhalf = lane >> 4, lmod = lane & 15;

  __shared__ _Float16 sK[64][34];        // [key][d]   (pairs along d for K^T frag)
  __shared__ _Float16 sVT[32][66];       // [d][key]   (pairs along key for V frag)
  __shared__ _Float16 sP[8][16][66];     // per-wave P staging [m][key]

  const int q0 = wid * 16;

  // Q fragment, loaded once (A-layout, dh = 32 = full K of one WMMA)
  Frag aq;
  {
    const int mrow = imin(q0 + lmod, Qn - 1);
    const float* qp = qh + ((size_t)b * Qn + mrow) * 256 + h * 32;
#pragma unroll
    for (int j = 0; j < 8; ++j) {
      const int k0 = 16 * (j >> 2) + 8 * lhalf + 2 * (j & 3);
      h2 t = {(_Float16)qp[k0], (_Float16)qp[k0 + 1]};
      aq.p[j] = t;
    }
  }

  v8f oacc0 = v8f_zero(), oacc1 = v8f_zero();
  float rowM[8], rowL[8];
#pragma unroll
  for (int r = 0; r < 8; ++r) { rowM[r] = -3e38f; rowL[r] = 0.f; }

  const int stKey = tid >> 2, stD = (tid & 3) * 8;
  const int nchunk = (Skv + 63) >> 6;

  for (int ch = 0; ch < nchunk; ++ch) {
    const int kc = ch * 64;
    __syncthreads();
    {  // stage K chunk [64x32] and V chunk transposed [32x64], f32 -> f16
      const int gk = kc + stKey;
      const bool valid = gk < Skv;
      const size_t kbase = ((size_t)b * Skv + (valid ? gk : (Skv - 1))) * 256 + h * 32 + stD;
      const float* kp = kh + kbase;
      const float* vp = vh + kbase;
#pragma unroll
      for (int u = 0; u < 8; ++u) {
        sK[stKey][stD + u]  = valid ? (_Float16)kp[u] : (_Float16)0.f;
        sVT[stD + u][stKey] = valid ? (_Float16)vp[u] : (_Float16)0.f;
      }
    }
    __syncthreads();

    // ---- scores: S = Q @ K^T for 4 key tiles of 16
    v8f sacc[4];
#pragma unroll
    for (int t = 0; t < 4; ++t) {
      Frag bk;
#pragma unroll
      for (int j = 0; j < 8; ++j)
        bk.p[j] = *(const h2*)&sK[t * 16 + lmod][16 * (j >> 2) + 8 * lhalf + 2 * (j & 3)];
      sacc[t] = __builtin_amdgcn_wmma_f32_16x16x32_f16(
          false, aq.v, false, bk.v, (short)0, v8f_zero(), false, false);
    }

    // ---- scale + rel bias + mask bias + key-range mask
    float sv[4][8], cmax[8];
#pragma unroll
    for (int r = 0; r < 8; ++r) cmax[r] = -3e38f;
#pragma unroll
    for (int t = 0; t < 4; ++t) {
      const int key = kc + t * 16 + lmod;
#pragma unroll
      for (int r = 0; r < 8; ++r) {
        const int qrow = imin(q0 + r + 8 * lhalf, Qn - 1);
        float xv;
        if (key < Skv) {
          xv = sacc[t][r] * scale + rel[((size_t)h * Qn + qrow) * Skv + key];
          if (mbias) xv += mbias[((size_t)b * Qn + qrow) * Skv + key];
        } else {
          xv = -1e30f;
        }
        sv[t][r] = xv;
        cmax[r] = fmaxf(cmax[r], xv);
      }
    }

    // ---- online softmax: row reductions across 16-lane halves (wave32)
#pragma unroll
    for (int r = 0; r < 8; ++r) {
#pragma unroll
      for (int m = 8; m >= 1; m >>= 1)
        cmax[r] = fmaxf(cmax[r], __shfl_xor(cmax[r], m, 32));
      const float newM = fmaxf(rowM[r], cmax[r]);
      const float alpha = __expf(rowM[r] - newM);
      rowM[r] = newM;
      rowL[r] *= alpha;
      oacc0[r] *= alpha;
      oacc1[r] *= alpha;
    }
    float psum[8];
#pragma unroll
    for (int r = 0; r < 8; ++r) psum[r] = 0.f;
#pragma unroll
    for (int t = 0; t < 4; ++t) {
#pragma unroll
      for (int r = 0; r < 8; ++r) {
        const float p = __expf(sv[t][r] - rowM[r]);
        psum[r] += p;
        sP[wid][r + 8 * lhalf][t * 16 + lmod] = (_Float16)p;  // C-layout -> LDS
      }
    }
#pragma unroll
    for (int r = 0; r < 8; ++r) {
#pragma unroll
      for (int m = 8; m >= 1; m >>= 1) psum[r] += __shfl_xor(psum[r], m, 32);
      rowL[r] += psum[r];
    }

    // ---- O += P @ V  (K = 64 keys -> two 32-key WMMA steps, N = 32 -> 2 tiles)
#pragma unroll
    for (int sub = 0; sub < 2; ++sub) {
      Frag ap;
#pragma unroll
      for (int j = 0; j < 8; ++j) {
        const int k0 = 32 * sub + 16 * (j >> 2) + 8 * lhalf + 2 * (j & 3);
        ap.p[j] = *(const h2*)&sP[wid][lmod][k0];
      }
      Frag bv0, bv1;
#pragma unroll
      for (int j = 0; j < 8; ++j) {
        const int kk = 32 * sub + 16 * (j >> 2) + 8 * lhalf + 2 * (j & 3);
        bv0.p[j] = *(const h2*)&sVT[lmod][kk];
        bv1.p[j] = *(const h2*)&sVT[16 + lmod][kk];
      }
      oacc0 = __builtin_amdgcn_wmma_f32_16x16x32_f16(
          false, ap.v, false, bv0.v, (short)0, oacc0, false, false);
      oacc1 = __builtin_amdgcn_wmma_f32_16x16x32_f16(
          false, ap.v, false, bv1.v, (short)0, oacc1, false, false);
    }
  }

  // ---- epilogue: normalize and scatter per-head columns
#pragma unroll
  for (int r = 0; r < 8; ++r) {
    const int qrow = q0 + r + 8 * lhalf;
    if (qrow < Qn) {
      const float invL = 1.f / rowL[r];
      float* op = outp + ((size_t)b * Qn + qrow) * 256 + h * 32;
      op[lmod]      = oacc0[r] * invL;
      op[16 + lmod] = oacc1[r] * invL;
    }
  }
}

// ---------------------------------------------------------------------------
// LayerNorm (optionally with residual add):  out = LN(x [+ res]) * g + b
// one block per row, 256 threads == C.
// ---------------------------------------------------------------------------
__global__ __launch_bounds__(256) void ln_kernel(
    const float* __restrict__ xin, const float* __restrict__ res,
    const float* __restrict__ gamma, const float* __restrict__ beta,
    float* __restrict__ out)
{
  const int row = blockIdx.x, tid = threadIdx.x;
  const size_t base = (size_t)row * 256 + tid;
  float v = xin[base];
  if (res) v += res[base];
  float s = v, s2 = v * v;
#pragma unroll
  for (int m = 16; m >= 1; m >>= 1) {
    s += __shfl_xor(s, m, 32);
    s2 += __shfl_xor(s2, m, 32);
  }
  __shared__ float ss[8], ss2[8];
  if ((tid & 31) == 0) { ss[tid >> 5] = s; ss2[tid >> 5] = s2; }
  __syncthreads();
  float ts = 0.f, ts2 = 0.f;
#pragma unroll
  for (int w = 0; w < 8; ++w) { ts += ss[w]; ts2 += ss2[w]; }
  const float mu = ts * (1.f / 256.f);
  const float var = ts2 * (1.f / 256.f) - mu * mu;
  const float inv = rsqrtf(var + 1e-5f);
  out[base] = (v - mu) * inv * gamma[tid] + beta[tid];
}

// ---------------------------------------------------------------------------
// Mask2Former mask bias:  blocked = sigmoid(mp)<0.5  <=>  mp<0,
// un-block rows that would be fully blocked.  One block per (b,q) row, S=1024.
// ---------------------------------------------------------------------------
__global__ __launch_bounds__(256) void maskbias_kernel(
    const float* __restrict__ mp, float* __restrict__ bias)
{
  const int row = blockIdx.x, tid = threadIdx.x;
  const float* rp = mp + (size_t)row * 1024;
  float* op = bias + (size_t)row * 1024;
  int blk[4]; int cnt = 0;
#pragma unroll
  for (int u = 0; u < 4; ++u) {
    blk[u] = (rp[tid + 256 * u] < 0.f) ? 1 : 0;
    cnt += blk[u];
  }
#pragma unroll
  for (int m = 16; m >= 1; m >>= 1) cnt += __shfl_xor(cnt, m, 32);
  __shared__ int sc[8];
  if ((tid & 31) == 0) sc[tid >> 5] = cnt;
  __syncthreads();
  int tot = 0;
#pragma unroll
  for (int w = 0; w < 8; ++w) tot += sc[w];
  const bool all = (tot == 1024);
#pragma unroll
  for (int u = 0; u < 4; ++u)
    op[tid + 256 * u] = (blk[u] && !all) ? -1e9f : 0.f;
}

// ---------------------------------------------------------------------------
// Elementwise helpers
// ---------------------------------------------------------------------------
__global__ void add_bcast_kernel(const float* __restrict__ a, const float* __restrict__ b,
                                 int bmod, float* __restrict__ o, int n)
{
  const int i = blockIdx.x * 256 + threadIdx.x;
  if (i < n) o[i] = a[i] + b[i % bmod];
}
__global__ void copy_bcast_kernel(const float* __restrict__ s, int smod,
                                  float* __restrict__ o, int n)
{
  const int i = blockIdx.x * 256 + threadIdx.x;
  if (i < n) o[i] = s[i % smod];
}

// ---------------------------------------------------------------------------
// Host orchestration
// ---------------------------------------------------------------------------
namespace {
enum : int { L_ = 9, H_ = 8, C_ = 256, F_ = 2048, Q_ = 100, B_ = 16, S_ = 1024,
             HW_ = 1024, MD_ = 256 };
const float SCALE_ = 0.17677669529663687f;  // 1/sqrt(32)

inline void launch_gemm(hipStream_t st, const float* A, int lda, long long bsA,
                        const float* W, long long bsW, const float* bias,
                        float* O, int ldo, long long bsO,
                        int M, int N, int K, int flags, int batch)
{
  dim3 grid((unsigned)((N + 127) / 128), (unsigned)((M + 127) / 128), (unsigned)batch);
  gemm_kernel<<<grid, 256, 0, st>>>(A, lda, bsA, W, bsW, bias, O, ldo, bsO, M, N, K, flags);
}

inline void predict_mask(hipStream_t st, const float* outb,
                         const float* dec_g, const float* dec_b,
                         const float* w1, const float* b1,
                         const float* w2, const float* b2,
                         const float* w3, const float* b3,
                         const float* mf, float* e1, float* e2, float* mp)
{
  ln_kernel<<<B_ * Q_, 256, 0, st>>>(outb, nullptr, dec_g, dec_b, e1);
  launch_gemm(st, e1, C_, 0, w1, 0, b1, e2, C_, 0, B_ * Q_, C_, C_, 1, 1);
  launch_gemm(st, e2, C_, 0, w2, 0, b2, e1, C_, 0, B_ * Q_, C_, C_, 1, 1);
  launch_gemm(st, e1, C_, 0, w3, 0, b3, e2, MD_, 0, B_ * Q_, MD_, C_, 0, 1);
  // per-batch einsum: e[b] (100x256) @ mask_features[b] (256x1024) -> mp[b]
  launch_gemm(st, e2, MD_, (long long)Q_ * MD_, mf, (long long)MD_ * HW_, nullptr,
              mp, HW_, (long long)Q_ * HW_, Q_, HW_, MD_, 0, B_);
}
}  // namespace

extern "C" void kernel_launch(void* const* d_in, const int* in_sizes, int n_in,
                              void* d_out, int out_size, void* d_ws, size_t ws_size,
                              hipStream_t stream)
{
  (void)in_sizes; (void)n_in; (void)out_size; (void)ws_size;
  const float* x      = (const float*)d_in[0];
  const float* pos    = (const float*)d_in[1];
  const float* mf     = (const float*)d_in[2];
  const float* qf     = (const float*)d_in[3];
  const float* qe     = (const float*)d_in[4];
  const float* ca_wq  = (const float*)d_in[5];
  const float* ca_wk  = (const float*)d_in[6];
  const float* ca_wv  = (const float*)d_in[7];
  const float* ca_wo  = (const float*)d_in[8];
  const float* ca_rel = (const float*)d_in[9];
  const float* ca_g   = (const float*)d_in[10];
  const float* ca_b   = (const float*)d_in[11];
  const float* sa_wq  = (const float*)d_in[12];
  const float* sa_wk  = (const float*)d_in[13];
  const float* sa_wv  = (const float*)d_in[14];
  const float* sa_wo  = (const float*)d_in[15];
  const float* sa_rel = (const float*)d_in[16];
  const float* sa_g   = (const float*)d_in[17];
  const float* sa_b   = (const float*)d_in[18];
  const float* ff_w1  = (const float*)d_in[19];
  const float* ff_b1  = (const float*)d_in[20];
  const float* ff_w2  = (const float*)d_in[21];
  const float* ff_b2  = (const float*)d_in[22];
  const float* ff_g   = (const float*)d_in[23];
  const float* ff_b   = (const float*)d_in[24];
  const float* dec_g  = (const float*)d_in[25];
  const float* dec_b  = (const float*)d_in[26];
  const float* me_w1  = (const float*)d_in[27];
  const float* me_b1  = (const float*)d_in[28];
  const float* me_w2  = (const float*)d_in[29];
  const float* me_b2  = (const float*)d_in[30];
  const float* me_w3  = (const float*)d_in[31];
  const float* me_b3  = (const float*)d_in[32];

  float* mp = (float*)d_out;  // [B,Q,32,32] lives in d_out throughout

  // workspace layout (floats)
  float* ws = (float*)d_ws;
  size_t o = 0;
  const size_t BSC = (size_t)B_ * S_ * C_;   // 4,194,304
  const size_t BQC = (size_t)B_ * Q_ * C_;   //   409,600
  float* kpos  = ws + o; o += BSC;
  float* khb   = ws + o; o += BSC;
  float* vhb   = ws + o; o += BSC;
  float* qkb   = ws + o; o += BQC;
  float* qhb   = ws + o; o += BQC;
  float* attnb = ws + o; o += BQC;
  float* t2b   = ws + o; o += BQC;
  float* outb  = ws + o; o += BQC;
  float* h1b   = ws + o; o += (size_t)B_ * Q_ * F_;
  float* e1b   = ws + o; o += BQC;
  float* e2b   = ws + o; o += BQC;
  float* mbb   = ws + o; o += (size_t)B_ * Q_ * S_;

  const int nBSC = (int)BSC, nBQC = (int)BQC;

  // kpos = x + pos ; out = broadcast(query_feat)
  add_bcast_kernel<<<(nBSC + 255) / 256, 256, 0, stream>>>(x, pos, nBSC, kpos, nBSC);
  copy_bcast_kernel<<<(nBQC + 255) / 256, 256, 0, stream>>>(qf, Q_ * C_, outb, nBQC);

  // initial mask prediction (gates layer-0 cross-attention)
  predict_mask(stream, outb, dec_g, dec_b, me_w1, me_b1, me_w2, me_b2,
               me_w3, me_b3, mf, e1b, e2b, mp);

  for (int l = 0; l < L_; ++l) {
    const size_t wofs = (size_t)l * C_ * C_;
    const size_t cofs = (size_t)l * C_;

    // ---- mask bias from previous mp
    maskbias_kernel<<<B_ * Q_, 256, 0, stream>>>(mp, mbb);

    // ---- cross-attention
    add_bcast_kernel<<<(nBQC + 255) / 256, 256, 0, stream>>>(outb, qe, Q_ * C_, qkb, nBQC);
    launch_gemm(stream, qkb, C_, 0, ca_wq + wofs, 0, nullptr, qhb, C_, 0, B_ * Q_, C_, C_, 0, 1);
    launch_gemm(stream, kpos, C_, 0, ca_wk + wofs, 0, nullptr, khb, C_, 0, B_ * S_, C_, C_, 0, 1);
    launch_gemm(stream, x,    C_, 0, ca_wv + wofs, 0, nullptr, vhb, C_, 0, B_ * S_, C_, C_, 0, 1);
    attn_kernel<<<dim3(H_, B_), 256, 0, stream>>>(
        qhb, khb, vhb, ca_rel + (size_t)l * H_ * Q_ * S_, mbb, attnb, Q_, S_, SCALE_);
    launch_gemm(stream, attnb, C_, 0, ca_wo + wofs, 0, nullptr, t2b, C_, 0, B_ * Q_, C_, C_, 0, 1);
    ln_kernel<<<B_ * Q_, 256, 0, stream>>>(outb, t2b, ca_g + cofs, ca_b + cofs, outb);

    // ---- self-attention
    add_bcast_kernel<<<(nBQC + 255) / 256, 256, 0, stream>>>(outb, qe, Q_ * C_, qkb, nBQC);
    launch_gemm(stream, qkb,  C_, 0, sa_wq + wofs, 0, nullptr, qhb, C_, 0, B_ * Q_, C_, C_, 0, 1);
    launch_gemm(stream, qkb,  C_, 0, sa_wk + wofs, 0, nullptr, khb, C_, 0, B_ * Q_, C_, C_, 0, 1);
    launch_gemm(stream, outb, C_, 0, sa_wv + wofs, 0, nullptr, vhb, C_, 0, B_ * Q_, C_, C_, 0, 1);
    attn_kernel<<<dim3(H_, B_), 256, 0, stream>>>(
        qhb, khb, vhb, sa_rel + (size_t)l * H_ * Q_ * Q_, nullptr, attnb, Q_, Q_, SCALE_);
    launch_gemm(stream, attnb, C_, 0, sa_wo + wofs, 0, nullptr, t2b, C_, 0, B_ * Q_, C_, C_, 0, 1);
    ln_kernel<<<B_ * Q_, 256, 0, stream>>>(outb, t2b, sa_g + cofs, sa_b + cofs, outb);

    // ---- FFN
    launch_gemm(stream, outb, C_, 0, ff_w1 + (size_t)l * C_ * F_, 0, ff_b1 + (size_t)l * F_,
                h1b, F_, 0, B_ * Q_, F_, C_, 1, 1);
    launch_gemm(stream, h1b, F_, 0, ff_w2 + (size_t)l * F_ * C_, 0, ff_b2 + cofs,
                t2b, C_, 0, B_ * Q_, C_, F_, 0, 1);
    ln_kernel<<<B_ * Q_, 256, 0, stream>>>(outb, t2b, ff_g + cofs, ff_b + cofs, outb);

    // ---- mask prediction for next layer (final one is the output)
    predict_mask(stream, outb, dec_g, dec_b, me_w1, me_b1, me_w2, me_b2,
                 me_w3, me_b3, mf, e1b, e2b, mp);
  }
}
